// S5Block_28939489640966
// MI455X (gfx1250) — matrix operand order repaired and checked by reading, not compiled, not run on hardware
//
#include <hip/hip_runtime.h>
#include <hip/hip_bf16.h>
#include <math.h>

typedef __attribute__((ext_vector_type(16))) __bf16 v16bf;
typedef __attribute__((ext_vector_type(8)))  float  v8f;
typedef __attribute__((ext_vector_type(4)))  int    v4i;

#define BM 128
#define BN 128
#define BK 32
#define LDST 40   // padded LDS row stride (elements): bank-conflict-free fragment reads

// ---- CDNA5 async Global->LDS path (probe via __has_builtin, safe fallback) ----
#if defined(__HIP_DEVICE_COMPILE__)
#if __has_builtin(__builtin_amdgcn_global_load_async_to_lds_b128)
#define HAVE_ASYNC_B128 1
#endif
#if __has_builtin(__builtin_amdgcn_s_wait_asynccnt)
#define HAVE_WAIT_ASYNC 1
#endif
#endif

#ifdef HAVE_ASYNC_B128
// builtin signature (from clang diagnostic): (int4 AS1*, int4 AS3*, imm, imm)
typedef __attribute__((address_space(1))) v4i GV4;
typedef __attribute__((address_space(3))) v4i LV4;

// one 16-byte DMA: global -> LDS, no VGPR data movement (ASYNCcnt-tracked)
__device__ __forceinline__ void async_ld16(const __bf16* g, const __bf16* l) {
    GV4* gp = (GV4*)(uintptr_t)g;
    LV4* lp = (LV4*)(uint32_t)(uintptr_t)l;   // low 32 bits = LDS byte address
    __builtin_amdgcn_global_load_async_to_lds_b128(gp, lp, 0, 0);
}

__device__ __forceinline__ void wait_async0() {
#ifdef HAVE_WAIT_ASYNC
    __builtin_amdgcn_s_wait_asynccnt(0);
#else
    asm volatile("s_wait_asynccnt 0x0" ::: "memory");
#endif
}
#endif

// ---------------------------------------------------------------------------
// Generic bf16 WMMA GEMM:  out[M][N] = A[M][K] * Bt[N][K]^T  (+ fused epilogue)
// MODE 0: plain f32 store
// MODE 1: + xn*D + residual(x), xn recomputed from (x, mu, rstd, gamma, beta)
// MODE 2: bias + exact GELU, store bf16
// MODE 3: bias + addend(y), store f32
// ---------------------------------------------------------------------------
template <int MODE>
__global__ __launch_bounds__(256) void gemm_wmma_bf16(
    const __bf16* __restrict__ A, const __bf16* __restrict__ Bt,
    int M, int N, int K,
    float* __restrict__ outF, __bf16* __restrict__ outH,
    const float* __restrict__ bias,
    const float* __restrict__ addend,
    const float* __restrict__ mu, const float* __restrict__ rstd,
    const float* __restrict__ gamma, const float* __restrict__ beta,
    const float* __restrict__ Dvec)
{
    __shared__ __bf16 As[BM * LDST];
    __shared__ __bf16 Bs[BN * LDST];

    const int tid  = threadIdx.x;
    const int lane = tid & 31;
    const int wave = tid >> 5;
    const int m_wave = (wave & 3) * 32;   // 4 waves along M
    const int n_wave = (wave >> 2) * 64;  // 2 waves along N
    const int mrow = lane & 15;
    const int h    = lane >> 4;

    const int blockN0 = blockIdx.x * BN;
    const int blockM0 = blockIdx.y * BM;

    // global->LDS staging: thread handles half a row (16 bf16 = 2x b128)
    const int ldr = tid >> 1;
    const int ldc = (tid & 1) * 16;

    v8f acc[2][4] = {};

#ifdef HAVE_ASYNC_B128
    // Padding rows (row >= M / N) are k-invariant: zero them ONCE, then the
    // async DMAs only ever overwrite valid rows inside the K loop.
    {
        const uint4 z = {0, 0, 0, 0};
        if (blockM0 + ldr >= M) {
            uint4* d = (uint4*)(As + ldr * LDST + ldc); d[0] = z; d[1] = z;
        }
        if (blockN0 + ldr >= N) {
            uint4* d = (uint4*)(Bs + ldr * LDST + ldc); d[0] = z; d[1] = z;
        }
    }
#endif

    for (int k0 = 0; k0 < K; k0 += BK) {
#ifdef HAVE_ASYNC_B128
        {   // DMA both tiles straight into LDS (no VGPR staging, ASYNCcnt)
            const int gr = blockM0 + ldr;
            if (gr < M) {
                const __bf16* g = A + (size_t)gr * K + k0 + ldc;
                const __bf16* l = As + ldr * LDST + ldc;
                async_ld16(g, l);
                async_ld16(g + 8, l + 8);
            }
            const int gn = blockN0 + ldr;
            if (gn < N) {
                const __bf16* g = Bt + (size_t)gn * K + k0 + ldc;
                const __bf16* l = Bs + ldr * LDST + ldc;
                async_ld16(g, l);
                async_ld16(g + 8, l + 8);
            }
            wait_async0();   // per-wave async ops retire in order; 0 => tile done
        }
#else
        {   // fallback: register staging
            uint4 q0 = {0,0,0,0}, q1 = {0,0,0,0};
            const int gr = blockM0 + ldr;
            if (gr < M) {
                const uint4* src = (const uint4*)(A + (size_t)gr * K + k0 + ldc);
                q0 = src[0]; q1 = src[1];
            }
            uint4* dst = (uint4*)(As + ldr * LDST + ldc);
            dst[0] = q0; dst[1] = q1;
        }
        {
            uint4 q0 = {0,0,0,0}, q1 = {0,0,0,0};
            const int gn = blockN0 + ldr;
            if (gn < N) {
                const uint4* src = (const uint4*)(Bt + (size_t)gn * K + k0 + ldc);
                q0 = src[0]; q1 = src[1];
            }
            uint4* dst = (uint4*)(Bs + ldr * LDST + ldc);
            dst[0] = q0; dst[1] = q1;
        }
#endif
        __syncthreads();

        // prefetch next K tile into cache while this one computes
        if (k0 + BK < K) {
            const int gr = blockM0 + ldr;
            if (gr < M) __builtin_prefetch(A + (size_t)gr * K + k0 + BK + ldc, 0, 1);
            const int gn = blockN0 + ldr;
            if (gn < N) __builtin_prefetch(Bt + (size_t)gn * K + k0 + BK + ldc, 0, 1);
        }

        union FragU { uint4 q[2]; v16bf v; };
        FragU a[2], b[4];
        // A-fragment ISA layout: lane m=lane&15; elem e -> k = (e&8)*2 + 8h + (e&7)
        for (int mt = 0; mt < 2; ++mt) {
            const __bf16* p = As + (m_wave + mt * 16 + mrow) * LDST;
            a[mt].q[0] = *(const uint4*)(p + h * 8);
            a[mt].q[1] = *(const uint4*)(p + 16 + h * 8);
        }
        // B-fragment ISA layout: lane n=lane&15; elem e -> k = (e&8)*2 + 8h + (e&7)
        for (int nt = 0; nt < 4; ++nt) {
            const __bf16* p = Bs + (n_wave + nt * 16 + mrow) * LDST;
            b[nt].q[0] = *(const uint4*)(p + h * 8);
            b[nt].q[1] = *(const uint4*)(p + 16 + h * 8);
        }

        for (int mt = 0; mt < 2; ++mt)
            for (int nt = 0; nt < 4; ++nt)
                acc[mt][nt] = __builtin_amdgcn_wmma_f32_16x16x32_bf16(
                    false, a[mt].v, false, b[nt].v, (short)0, acc[mt][nt],
                    false, false);
        __syncthreads();
    }

    // C/D layout: VGPR v, lanes0-15 -> M=v, lanes16-31 -> M=8+v, N=lane&15
    for (int mt = 0; mt < 2; ++mt) {
        for (int nt = 0; nt < 4; ++nt) {
            const int col = blockN0 + n_wave + nt * 16 + mrow;
            if (col >= N) continue;
            for (int v = 0; v < 8; ++v) {
                const int row = blockM0 + m_wave + mt * 16 + 8 * h + v;
                if (row >= M) continue;
                float val = acc[mt][nt][v];
                const size_t idx = (size_t)row * N + col;
                if (MODE == 0) {
                    outF[idx] = val;
                } else if (MODE == 1) {
                    const float xv = addend[idx];
                    const float xn = (xv - mu[row]) * rstd[row] * gamma[col] + beta[col];
                    outF[idx] = val + xn * Dvec[col] + xv;
                } else if (MODE == 2) {
                    const float t = val + bias[col];
                    const float g = 0.5f * t * (1.0f + erff(t * 0.70710678118654752f));
                    outH[idx] = (__bf16)g;
                } else {
                    outF[idx] = val + bias[col] + addend[idx];
                }
            }
        }
    }
}

// ---------------------------------------------------------------------------
// LayerNorm over D=512, writes bf16 normalized output (+ optional mu/rstd)
// ---------------------------------------------------------------------------
__global__ __launch_bounds__(256) void ln_kernel(
    const float* __restrict__ in, const float* __restrict__ gamma,
    const float* __restrict__ beta, __bf16* __restrict__ outH,
    float* __restrict__ muOut, float* __restrict__ rstdOut)
{
    __shared__ float red[256];
    const int row = blockIdx.x;
    const int t = threadIdx.x;
    const float* r = in + (size_t)row * 512;
    const float v0 = r[t], v1 = r[t + 256];

    red[t] = v0 + v1;
    __syncthreads();
    for (int s = 128; s > 0; s >>= 1) { if (t < s) red[t] += red[t + s]; __syncthreads(); }
    const float mu = red[0] * (1.0f / 512.0f);
    __syncthreads();

    const float d0 = v0 - mu, d1 = v1 - mu;
    red[t] = d0 * d0 + d1 * d1;
    __syncthreads();
    for (int s = 128; s > 0; s >>= 1) { if (t < s) red[t] += red[t + s]; __syncthreads(); }
    const float rstd = rsqrtf(red[0] * (1.0f / 512.0f) + 1e-5f);

    outH[(size_t)row * 512 + t]       = (__bf16)(d0 * rstd * gamma[t] + beta[t]);
    outH[(size_t)row * 512 + t + 256] = (__bf16)(d1 * rstd * gamma[t + 256] + beta[t + 256]);
    if (muOut && t == 0) { muOut[row] = mu; rstdOut[row] = rstd; }
}

// ---------------------------------------------------------------------------
// Chunked parallel scan (A_bar is time-invariant)
// ---------------------------------------------------------------------------
__device__ __forceinline__ float abar_of(const float* logA, const float* logdt, int s) {
    return __expf(-__expf(logA[s]) * __expf(logdt[0]));
}

__global__ void scan_chunk(const float* __restrict__ xB, const float* __restrict__ logA,
                           const float* __restrict__ logdt,
                           float* __restrict__ hsL, float* __restrict__ carry)
{
    const int b = blockIdx.x, c = blockIdx.y, s = threadIdx.x; // 64 threads
    const float Ab = abar_of(logA, logdt, s);
    const size_t base = ((size_t)b * 4096 + (size_t)c * 256) * 64 + s;
    float hcur = 0.0f;
    for (int t = 0; t < 256; ++t) {
        hcur = Ab * hcur + xB[base + (size_t)t * 64];
        hsL[base + (size_t)t * 64] = hcur;
    }
    carry[((size_t)b * 16 + c) * 64 + s] = hcur;
}

__global__ void scan_carry(const float* __restrict__ carry, const float* __restrict__ logA,
                           const float* __restrict__ logdt, float* __restrict__ hstart)
{
    const int b = blockIdx.x, s = threadIdx.x;
    const float Ab = abar_of(logA, logdt, s);
    float p = Ab;
    for (int i = 0; i < 8; ++i) p *= p;   // Ab^256 via 8 squarings
    float hcur = 0.0f;
    hstart[((size_t)b * 16 + 0) * 64 + s] = 0.0f;
    for (int c = 1; c < 16; ++c) {
        hcur = p * hcur + carry[((size_t)b * 16 + c - 1) * 64 + s];
        hstart[((size_t)b * 16 + c) * 64 + s] = hcur;
    }
}

__global__ void scan_fix(const float* __restrict__ hsL, const float* __restrict__ hstart,
                         const float* __restrict__ logA, const float* __restrict__ logdt,
                         __bf16* __restrict__ hsH)
{
    const int b = blockIdx.x, c = blockIdx.y, s = threadIdx.x;
    const float Ab = abar_of(logA, logdt, s);
    const float h0 = hstart[((size_t)b * 16 + c) * 64 + s];
    const size_t base = ((size_t)b * 4096 + (size_t)c * 256) * 64 + s;
    float pw = Ab;
    for (int t = 0; t < 256; ++t) {
        hsH[base + (size_t)t * 64] = (__bf16)(hsL[base + (size_t)t * 64] + pw * h0);
        pw *= Ab;
    }
}

// Weight convert + transpose: Wt[n][k] = (bf16) W[k][n]   (W is [K][N])
__global__ void wconv(const float* __restrict__ W, __bf16* __restrict__ Wt, int K, int N)
{
    const long i = (long)blockIdx.x * 256 + threadIdx.x;
    if (i < (long)K * N) {
        const int n = (int)(i / K), k = (int)(i % K);
        Wt[i] = (__bf16)W[(size_t)k * N + n];
    }
}

// ---------------------------------------------------------------------------
extern "C" void kernel_launch(void* const* d_in, const int* in_sizes, int n_in,
                              void* d_out, int out_size, void* d_ws, size_t ws_size,
                              hipStream_t stream)
{
    const float* x     = (const float*)d_in[0];
    const float* logA  = (const float*)d_in[1];
    const float* Bw    = (const float*)d_in[2];
    const float* Cw    = (const float*)d_in[3];
    const float* Dv    = (const float*)d_in[4];
    const float* logdt = (const float*)d_in[5];
    const float* gamma = (const float*)d_in[6];
    const float* beta  = (const float*)d_in[7];
    const float* W1    = (const float*)d_in[8];
    const float* b1    = (const float*)d_in[9];
    const float* W2    = (const float*)d_in[10];
    const float* b2    = (const float*)d_in[11];
    float* out = (float*)d_out;

    const int Dm = 512, S = 64, M = 8 * 4096;

    char* ws = (char*)d_ws;
    size_t off = 0;
    auto alloc = [&](size_t bytes) -> void* {
        const size_t a = (off + 255) & ~(size_t)255;
        off = a + bytes;
        return (void*)(ws + a);
    };

    float*  mu     = (float*) alloc((size_t)M * 4);
    float*  rstd   = (float*) alloc((size_t)M * 4);
    __bf16* xnH    = (__bf16*)alloc((size_t)M * Dm * 2);
    float*  xB     = (float*) alloc((size_t)M * S * 4);
    float*  hsL    = (float*) alloc((size_t)M * S * 4);
    float*  carry  = (float*) alloc((size_t)8 * 16 * 64 * 4);
    float*  hstart = (float*) alloc((size_t)8 * 16 * 64 * 4);
    __bf16* hsH    = (__bf16*)alloc((size_t)M * S * 2);
    float*  y      = (float*) alloc((size_t)M * Dm * 4);
    __bf16* ynH    = (__bf16*)alloc((size_t)M * Dm * 2);
    __bf16* h1H    = (__bf16*)alloc((size_t)M * 2 * Dm * 2);
    __bf16* BtH    = (__bf16*)alloc((size_t)Dm * S * 2);
    __bf16* CtH    = (__bf16*)alloc((size_t)S * Dm * 2);
    __bf16* W1tH   = (__bf16*)alloc((size_t)Dm * 2 * Dm * 2);
    __bf16* W2tH   = (__bf16*)alloc((size_t)2 * Dm * Dm * 2);
    (void)ws_size; (void)in_sizes; (void)n_in; (void)out_size;

    // weights -> bf16, transposed to [N][K]
    wconv<<<(Dm * S + 255) / 256, 256, 0, stream>>>(Bw, BtH, Dm, S);
    wconv<<<(S * Dm + 255) / 256, 256, 0, stream>>>(Cw, CtH, S, Dm);
    wconv<<<(Dm * 2 * Dm + 255) / 256, 256, 0, stream>>>(W1, W1tH, Dm, 2 * Dm);
    wconv<<<(2 * Dm * Dm + 255) / 256, 256, 0, stream>>>(W2, W2tH, 2 * Dm, Dm);

    // LN1 (saves mu/rstd so the D-term epilogue can recompute xn from x)
    ln_kernel<<<M, 256, 0, stream>>>(x, gamma, beta, xnH, mu, rstd);

    const dim3 blk(256);
    // xB = xn @ B      [M x 64], K=512
    gemm_wmma_bf16<0><<<dim3((S + 127) / 128, (M + 127) / 128), blk, 0, stream>>>(
        xnH, BtH, M, S, Dm, xB, nullptr,
        nullptr, nullptr, nullptr, nullptr, nullptr, nullptr, nullptr);

    // chunked scan: local scans -> carry prefix (A_bar^256) -> fixup (+bf16 cast)
    scan_chunk<<<dim3(8, 16), 64, 0, stream>>>(xB, logA, logdt, hsL, carry);
    scan_carry<<<8, 64, 0, stream>>>(carry, logA, logdt, hstart);
    scan_fix<<<dim3(8, 16), 64, 0, stream>>>(hsL, hstart, logA, logdt, hsH);

    // y = hs @ C + xn*D + x     [M x 512], K=64
    gemm_wmma_bf16<1><<<dim3(Dm / 128, (M + 127) / 128), blk, 0, stream>>>(
        hsH, CtH, M, Dm, S, y, nullptr,
        nullptr, x, mu, rstd, gamma, beta, Dv);

    // LN2
    ln_kernel<<<M, 256, 0, stream>>>(y, gamma, beta, ynH, nullptr, nullptr);

    // h1 = gelu(yn @ W1 + b1)   [M x 1024], K=512, bf16 out
    gemm_wmma_bf16<2><<<dim3(2 * Dm / 128, (M + 127) / 128), blk, 0, stream>>>(
        ynH, W1tH, M, 2 * Dm, Dm, nullptr, h1H,
        b1, nullptr, nullptr, nullptr, nullptr, nullptr, nullptr);

    // out = y + h1 @ W2 + b2    [M x 512], K=1024
    gemm_wmma_bf16<3><<<dim3(Dm / 128, (M + 127) / 128), blk, 0, stream>>>(
        h1H, W2tH, M, Dm, 2 * Dm, out, nullptr,
        b2, y, nullptr, nullptr, nullptr, nullptr, nullptr);
}
